// MultiHeadAttention_1022202216935
// MI455X (gfx1250) — matrix operand order repaired and checked by reading, not compiled, not run on hardware
//
#include <hip/hip_runtime.h>
#include <hip/hip_bf16.h>

typedef __bf16 bf16_t;
typedef __attribute__((ext_vector_type(16))) __bf16 v16bf;
typedef __attribute__((ext_vector_type(8)))  __bf16 v8bf;
typedef __attribute__((ext_vector_type(8)))  float  v8f;

#define SEQ    2048
#define BSZ    2
#define HEADS  16
#define DMODEL 1024
#define DK     64
#define NTOK   (SEQ * BSZ)   // 4096

union AF { v16bf v; v8bf h[2]; };

__device__ __forceinline__ v8f vzero8() {
  v8f z;
#pragma unroll
  for (int i = 0; i < 8; ++i) z[i] = 0.0f;
  return z;
}

__device__ __forceinline__ v8f wmma_bf16(v16bf a, v16bf b, v8f c) {
  // D = A(16x32 bf16) * B(32x16 bf16) + C(16x16 f32)
  return __builtin_amdgcn_wmma_f32_16x16x32_bf16(
      /*neg_a=*/false, a, /*neg_b=*/false, b,
      /*c_mod=*/(short)0, c, /*reuse_a=*/false, /*reuse_b=*/false);
}

// ---------------------------------------------------------------------------
// Kernel 0: fp32 -> bf16 bulk conversion (one pass). 8 elems/thread.
// ---------------------------------------------------------------------------
__global__ void __launch_bounds__(256)
cvt_bf16_kernel(const float* __restrict__ src, bf16_t* __restrict__ dst, int n)
{
  const int i8 = (blockIdx.x * 256 + threadIdx.x) * 8;
  if (i8 >= n) return;
  const float4 f0 = *(const float4*)(src + i8);
  const float4 f1 = *(const float4*)(src + i8 + 4);
  v8bf o;
  o[0] = (bf16_t)f0.x; o[1] = (bf16_t)f0.y;
  o[2] = (bf16_t)f0.z; o[3] = (bf16_t)f0.w;
  o[4] = (bf16_t)f1.x; o[5] = (bf16_t)f1.y;
  o[6] = (bf16_t)f1.z; o[7] = (bf16_t)f1.w;
  *(v8bf*)(dst + i8) = o;
}

// ---------------------------------------------------------------------------
// Kernel 1: fused Q/K/V projection  (q = x @ W^T + b), all-bf16 WMMA.
// One wave computes a 32(token) x 64(feature) tile: per K=32 step it loads
// 2 A-fragments + 4 B-fragments (12 x b128) and issues 8 v_wmma into 8
// f32 accumulators (~21 FLOP/byte of L2 traffic).
// Q,K stored [b,h,s,dk]; V stored TRANSPOSED [b,h,dk,s] for the PV matmul.
// ---------------------------------------------------------------------------
__global__ void __launch_bounds__(256)
qkv_proj_kernel(const bf16_t* __restrict__ xb, const bf16_t* __restrict__ Wb,
                const float* __restrict__ bq, const float* __restrict__ bk,
                const float* __restrict__ bv,
                bf16_t* __restrict__ Qb, bf16_t* __restrict__ Kb,
                bf16_t* __restrict__ Vt)
{
  const int lane = threadIdx.x & 31;
  const int half = lane >> 4;        // 0: low 16 lanes, 1: high 16 lanes
  const int l16  = lane & 15;

  const int wid = blockIdx.x * (blockDim.x >> 5) + (threadIdx.x >> 5);
  const int TO = DMODEL / 64;        // 16 feature tiles (64 wide)
  const int TT = NTOK / 32;          // 128 token tiles (32 tall)
  const int sel  = wid / (TT * TO);  // 0:Q 1:K 2:V
  const int rem  = wid - sel * TT * TO;
  const int tbase = (rem / TO) * 32;
  const int obase = (rem % TO) * 64;

  const bf16_t* W   = Wb + (size_t)sel * DMODEL * DMODEL;
  const float*  bia = (sel == 0) ? bq : (sel == 1) ? bk : bv;

  v8f acc[2][4];
#pragma unroll
  for (int rr = 0; rr < 2; ++rr)
#pragma unroll
    for (int cI = 0; cI < 4; ++cI) acc[rr][cI] = vzero8();

  // A: lane = row (token); e<8 -> K = 8*half+e ; e>=8 -> K = 8*half+16+(e-8)
  const bf16_t* xrow0 = xb + (tbase + l16) * DMODEL + 8 * half;
  const bf16_t* xrow1 = xrow0 + 16 * DMODEL;
  // B: lane = col (feature); same K order as A  (W row contiguous in d)
  const bf16_t* wrow = W + (obase + l16) * DMODEL + 8 * half;

#pragma unroll 2
  for (int ks = 0; ks < DMODEL; ks += 32) {
    AF a0, a1;
    a0.h[0] = *(const v8bf*)(xrow0 + ks);
    a0.h[1] = *(const v8bf*)(xrow0 + ks + 16);
    a1.h[0] = *(const v8bf*)(xrow1 + ks);
    a1.h[1] = *(const v8bf*)(xrow1 + ks + 16);
#pragma unroll
    for (int cI = 0; cI < 4; ++cI) {
      AF b;
      const bf16_t* p = wrow + (size_t)cI * 16 * DMODEL + ks;
      b.h[0] = *(const v8bf*)(p);
      b.h[1] = *(const v8bf*)(p + 16);
      acc[0][cI] = wmma_bf16(a0.v, b.v, acc[0][cI]);
      acc[1][cI] = wmma_bf16(a1.v, b.v, acc[1][cI]);
    }
  }

#pragma unroll
  for (int cI = 0; cI < 4; ++cI) {
    const int o  = obase + cI * 16 + l16;
    const int hh = o >> 6, dk = o & 63;
    const float bb = bia[o];
    if (sel == 2) {
      // V transposed: [b, h, dk, s]
#pragma unroll
      for (int rr = 0; rr < 2; ++rr)
#pragma unroll
        for (int r = 0; r < 8; ++r) {
          const int t = tbase + rr * 16 + r + 8 * half;  // token = s*BSZ + b
          const int s = t >> 1, bat = t & 1;
          Vt[(((bat * HEADS + hh) * DK) + dk) * SEQ + s] =
              (bf16_t)(acc[rr][cI][r] + bb);
        }
    } else {
      bf16_t* dst = (sel == 0) ? Qb : Kb;
#pragma unroll
      for (int rr = 0; rr < 2; ++rr)
#pragma unroll
        for (int r = 0; r < 8; ++r) {     // C layout: row = r+8*half, col = l16
          const int t = tbase + rr * 16 + r + 8 * half;
          const int s = t >> 1, bat = t & 1;
          dst[(((bat * HEADS + hh) * SEQ) + s) * DK + dk] =
              (bf16_t)(acc[rr][cI][r] + bb);
        }
    }
  }
}

// ---------------------------------------------------------------------------
// Kernel 2: causal flash attention with softmax-one (+1 denominator).
// One wave owns a 16-query tile of one (b,h); streams 32 keys per iteration.
// ---------------------------------------------------------------------------
__global__ void __launch_bounds__(256)
attn_kernel(const bf16_t* __restrict__ Qb, const bf16_t* __restrict__ Kb,
            const bf16_t* __restrict__ Vt, float* __restrict__ out)
{
  __shared__ __align__(16) bf16_t pbuf[8][16 * 32];  // per-wave P scratch (1 KB)

  const int lane = threadIdx.x & 31;
  const int half = lane >> 4;
  const int l16  = lane & 15;
  const int wv   = threadIdx.x >> 5;

  const int wid = blockIdx.x * 8 + wv;
  const int bh  = wid >> 7;          // 0..31  (b*16 + h)
  const int qt  = wid & 127;         // query tile index
  const int bat = bh >> 4, hh = bh & 15;

  const bf16_t* Qh = Qb + (size_t)bh * SEQ * DK;
  const bf16_t* Kh = Kb + (size_t)bh * SEQ * DK;
  const bf16_t* Vh = Vt + (size_t)bh * DK * SEQ;   // [dk][s]
  const int ibase = qt * 16;

  // Q in A layout, two K=32 chunks covering dk=64 (held for the whole loop)
  AF aq0, aq1;
  {
    const bf16_t* p = Qh + (ibase + l16) * DK + 8 * half;
    aq0.h[0] = *(const v8bf*)(p);
    aq0.h[1] = *(const v8bf*)(p + 16);
    aq1.h[0] = *(const v8bf*)(p + 32);
    aq1.h[1] = *(const v8bf*)(p + 48);
  }

  v8f acc[4];
#pragma unroll
  for (int cI = 0; cI < 4; ++cI) acc[cI] = vzero8();
  float m[8], l[8];
#pragma unroll
  for (int r = 0; r < 8; ++r) { m[r] = -__builtin_inff(); l[r] = 0.0f; }

  bf16_t* pw = pbuf[wv];
  const float scale = 0.125f;  // 1/sqrt(64)

  for (int jb = 0; jb <= ibase + 15; jb += 32) {
    // ---- scores: S = Q * K^T, two 16-key tiles ----
    v8f s0 = vzero8(), s1 = vzero8();
    {
      AF b0, b1;
      const bf16_t* p0 = Kh + (jb + l16) * DK + 8 * half;  // B: lane=key col
      b0.h[0] = *(const v8bf*)(p0);      b0.h[1] = *(const v8bf*)(p0 + 16);
      b1.h[0] = *(const v8bf*)(p0 + 32); b1.h[1] = *(const v8bf*)(p0 + 48);
      s0 = wmma_bf16(aq0.v, b0.v, s0);
      s0 = wmma_bf16(aq1.v, b1.v, s0);
      const bf16_t* p1 = Kh + (jb + 16 + l16) * DK + 8 * half;
      b0.h[0] = *(const v8bf*)(p1);      b0.h[1] = *(const v8bf*)(p1 + 16);
      b1.h[0] = *(const v8bf*)(p1 + 32); b1.h[1] = *(const v8bf*)(p1 + 48);
      s1 = wmma_bf16(aq0.v, b0.v, s1);
      s1 = wmma_bf16(aq1.v, b1.v, s1);
    }

    // ---- online softmax-one update (per C-layout row r + 8*half) ----
    float alpha[8];
    const int j0 = jb + l16, j1 = jb + 16 + l16;
#pragma unroll
    for (int r = 0; r < 8; ++r) {
      const int i = ibase + r + 8 * half;
      float v0 = (j0 <= i) ? s0[r] * scale : -__builtin_inff();
      float v1 = (j1 <= i) ? s1[r] * scale : -__builtin_inff();
      float mx = fmaxf(v0, v1);
#pragma unroll
      for (int d = 1; d < 16; d <<= 1)
        mx = fmaxf(mx, __shfl_xor(mx, d, 32));   // stays inside 16-lane half
      const float mn = fmaxf(m[r], mx);
      const float p0 = __expf(v0 - mn);
      const float p1 = __expf(v1 - mn);
      alpha[r] = __expf(m[r] - mn);
      float rs = p0 + p1;
#pragma unroll
      for (int d = 1; d < 16; d <<= 1)
        rs += __shfl_xor(rs, d, 32);
      l[r] = l[r] * alpha[r] + rs;
      m[r] = mn;
      // P (C layout) -> LDS row-major 16x32 bf16
      pw[(r + 8 * half) * 32 + l16]      = (bf16_t)p0;
      pw[(r + 8 * half) * 32 + 16 + l16] = (bf16_t)p1;
    }

    // rescale accumulator by alpha (rows match C layout)
#pragma unroll
    for (int cI = 0; cI < 4; ++cI)
#pragma unroll
      for (int r = 0; r < 8; ++r) acc[cI][r] *= alpha[r];

    // reload P in A layout from LDS (two 16B ds_load_b128 per lane)
    AF pa;
    {
      const bf16_t* p = pw + l16 * 32 + 8 * half;
      pa.h[0] = *(const v8bf*)(p);
      pa.h[1] = *(const v8bf*)(p + 16);
    }

    // ---- O += P(16x32) * V(32x64): V^T rows are contiguous in s ----
#pragma unroll
    for (int cI = 0; cI < 4; ++cI) {
      AF bvf;  // B[k=key][n=dk]: lane = dk col, keys in same order as P
      const bf16_t* p = Vh + (cI * 16 + l16) * SEQ + jb + 8 * half;
      bvf.h[0] = *(const v8bf*)(p);
      bvf.h[1] = *(const v8bf*)(p + 16);
      acc[cI] = wmma_bf16(pa.v, bvf.v, acc[cI]);
    }
  }

  // ---- epilogue: divide by (1 + l) and store fp32 output (S,B,D) ----
#pragma unroll
  for (int cI = 0; cI < 4; ++cI) {
#pragma unroll
    for (int r = 0; r < 8; ++r) {
      const int i = ibase + r + 8 * half;
      const float denom = 1.0f + l[r];
      out[(i * BSZ + bat) * DMODEL + hh * DK + cI * 16 + l16] =
          acc[cI][r] / denom;
    }
  }
}

// ---------------------------------------------------------------------------
extern "C" void kernel_launch(void* const* d_in, const int* in_sizes, int n_in,
                              void* d_out, int out_size, void* d_ws,
                              size_t ws_size, hipStream_t stream)
{
  (void)in_sizes; (void)n_in; (void)out_size; (void)ws_size;
  const float* x  = (const float*)d_in[0];
  const float* Wq = (const float*)d_in[1];
  const float* bq = (const float*)d_in[2];
  const float* Wk = (const float*)d_in[3];
  const float* bk = (const float*)d_in[4];
  const float* Wv = (const float*)d_in[5];
  const float* bv = (const float*)d_in[6];
  float* out = (float*)d_out;

  // Workspace layout (bf16): xb[4096x1024] 8MB | Wb[3x1024x1024] 6MB |
  //                          Qb 8MB | Kb 8MB | Vt 8MB   (total 38 MB)
  const size_t nX = (size_t)NTOK * DMODEL;        // 4 Mi elems
  const size_t nW = (size_t)DMODEL * DMODEL;      // 1 Mi elems
  const size_t perMat = (size_t)BSZ * HEADS * SEQ * DK;
  bf16_t* xb = (bf16_t*)d_ws;
  bf16_t* Wb = xb + nX;
  bf16_t* Qb = Wb + 3 * nW;
  bf16_t* Kb = Qb + perMat;
  bf16_t* Vt = Kb + perMat;

  // One-pass fp32 -> bf16 conversion
  cvt_bf16_kernel<<<(int)(nX / 8 / 256), 256, 0, stream>>>(x, xb, (int)nX);
  cvt_bf16_kernel<<<(int)(nW / 8 / 256), 256, 0, stream>>>(Wq, Wb + 0 * nW, (int)nW);
  cvt_bf16_kernel<<<(int)(nW / 8 / 256), 256, 0, stream>>>(Wk, Wb + 1 * nW, (int)nW);
  cvt_bf16_kernel<<<(int)(nW / 8 / 256), 256, 0, stream>>>(Wv, Wb + 2 * nW, (int)nW);

  const int projTiles = 3 * (NTOK / 32) * (DMODEL / 64);  // 6144 wave-tiles
  qkv_proj_kernel<<<projTiles / 8, 256, 0, stream>>>(xb, Wb, bq, bk, bv,
                                                     Qb, Kb, Vt);

  const int attnWaves = BSZ * HEADS * (SEQ / 16);          // 4096 wave-tiles
  attn_kernel<<<attnWaves / 8, 256, 0, stream>>>(Qb, Kb, Vt, out);
}
